// ECGModel_18081812316619
// MI455X (gfx1250) — matrix operand order repaired and checked by reading, not compiled, not run on hardware
//
#include <hip/hip_runtime.h>

// ---------------------------------------------------------------------------
// CDNA5 (gfx1250, wave32) implementation of the ECG masked-autoencoder fwd.
// All GEMM-shaped work runs through v_wmma_f32_16x16x32_bf16.
// ---------------------------------------------------------------------------

typedef __attribute__((ext_vector_type(16))) __bf16 v16bf;
typedef __attribute__((ext_vector_type(8)))  __bf16 v8bf;
typedef __attribute__((ext_vector_type(8)))  float  v8f;

#define SHUF16(x, y) __builtin_shufflevector((x), (y), 0,1,2,3,4,5,6,7,8,9,10,11,12,13,14,15)

__device__ __forceinline__ v8bf cvt8(float4 a, float4 b)
{
    v8bf r;
    r[0] = (__bf16)a.x; r[1] = (__bf16)a.y; r[2] = (__bf16)a.z; r[3] = (__bf16)a.w;
    r[4] = (__bf16)b.x; r[5] = (__bf16)b.y; r[6] = (__bf16)b.z; r[7] = (__bf16)b.w;
    return r;
}

// ---------------------------------------------------------------------------
// Tiled WMMA GEMM:  C = epi( alpha * A @ B  [*rowScale] [+bias] [+resid] )
//   A: (M x K) f32 row-major, leading dim lda, batch stride aB
//   B: if bIsNK: (N x K) row-major (PyTorch weight, dot-rows)   else (K x N)
//   C: (M x N) f32, leading dim ldc
// REQUIRES: M % 64 == 0, N % 64 == 0, K % 32 == 0 (true for every call here),
// and 16-byte aligned A/B base+lda (true: all dims are multiples of 16 floats).
// Block = 128 threads = 4 waves, block tile 64x64; each wave owns a 32x32
// quadrant = 2x2 WMMA tiles -> 4 v_wmma per wave per 32-wide k-step.
// Double-buffered LDS pipeline: global loads for tile k are issued into
// registers BEFORE the WMMAs consuming tile k-1, so s_wait_loadcnt sinks past
// the matrix ops; one barrier per k-step.
// Fragments follow the CDNA5 ISA 16-bit A/B/C VGPR layouts.
// ---------------------------------------------------------------------------
__global__ void __launch_bounds__(128)
gemm_bf16_wmma(const float* A, int lda, long long aB,
               const float* Bm, int ldb, long long bB, int bIsNK,
               const float* bias,
               const float* resid, int ldres, long long resB,
               const float* rowScale, long long rsB,
               float* C, int ldc, long long cB,
               int M, int Nout, int Kdim, int doRelu, float alpha)
{
    constexpr int LDST = 40;                         // bf16 elems per LDS row (80B, 16B aligned)
    __shared__ alignas(16) __bf16 As[2][64 * LDST];  // As[buf][m][k]
    __shared__ alignas(16) __bf16 Bs[2][64 * LDST];  // Bs[buf][n][k] (staged transposed)

    const int z = blockIdx.z;
    A  += (long long)z * aB;
    Bm += (long long)z * bB;
    C  += (long long)z * cB;
    if (resid)    resid    += (long long)z * resB;
    if (rowScale) rowScale += (long long)z * rsB;

    const int m0   = blockIdx.y * 64;
    const int n0   = blockIdx.x * 64;
    const int tid  = threadIdx.x;
    const int lane = tid & 31;
    const int wave = tid >> 5;
    const int wm   = (wave >> 1) * 32;   // wave quadrant row offset
    const int wn   = (wave & 1) * 32;    // wave quadrant col offset
    const int lr   = lane & 15;
    const int hi   = lane >> 4;

    const int srow = tid >> 1;           // 0..63  (staging row)
    const int scol = (tid & 1) * 16;     // 0 or 16 (staging k-base)

    v8f acc[2][2] = {};

    const float* aptr = A + (long long)(m0 + srow) * lda + scol;
    const float* bptr = bIsNK ? (Bm + (long long)(n0 + srow) * ldb + scol)
                              : (Bm + (n0 + srow));

    float4 ar[4], br[4];   // staged A / B (NK mode) registers
    float  bg[16];         // staged B (KxN gather mode) registers

    // issue global loads for k-tile k0 into registers (no LDS yet)
    auto load_tile = [&](int k0) {
        const float4* s = (const float4*)(aptr + k0);
        ar[0] = s[0]; ar[1] = s[1]; ar[2] = s[2]; ar[3] = s[3];
        if (bIsNK) {
            const float4* t = (const float4*)(bptr + k0);
            br[0] = t[0]; br[1] = t[1]; br[2] = t[2]; br[3] = t[3];
        } else {
#pragma unroll
            for (int j = 0; j < 16; ++j)
                bg[j] = bptr[(long long)(k0 + scol + j) * ldb];
        }
    };

    // convert staged registers to bf16 and store into LDS buffer `buf`
    auto store_tile = [&](int buf) {
        v8bf* da = (v8bf*)&As[buf][srow * LDST + scol];
        da[0] = cvt8(ar[0], ar[1]);
        da[1] = cvt8(ar[2], ar[3]);
        v8bf* db = (v8bf*)&Bs[buf][srow * LDST + scol];
        if (bIsNK) {
            db[0] = cvt8(br[0], br[1]);
            db[1] = cvt8(br[2], br[3]);
        } else {
            v8bf lo, hiv;
#pragma unroll
            for (int j = 0; j < 8; ++j) { lo[j] = (__bf16)bg[j]; hiv[j] = (__bf16)bg[8 + j]; }
            db[0] = lo;
            db[1] = hiv;
        }
    };

    // build fragments from LDS buffer `buf` and run the 2x2 WMMA tile
    // A (16-bit 16x32): lane(hi,lr) row M=lr, elems 0..7 -> K=hi*8+0..7,
    //                   elems 8..15 -> K=16+hi*8+0..7
    // B (32x16):        lane(hi,lr) col N=lr, same K mapping as A
    auto consume = [&](int buf) {
        v16bf af[2], bf[2];
#pragma unroll
        for (int mt = 0; mt < 2; ++mt) {
            const v8bf* p = (const v8bf*)&As[buf][(wm + mt * 16 + lr) * LDST];
            af[mt] = SHUF16(p[hi], p[2 + hi]);
        }
#pragma unroll
        for (int nt = 0; nt < 2; ++nt) {
            const v8bf* p = (const v8bf*)&Bs[buf][(wn + nt * 16 + lr) * LDST];
            bf[nt] = SHUF16(p[hi], p[2 + hi]);
        }
#pragma unroll
        for (int mt = 0; mt < 2; ++mt)
#pragma unroll
            for (int nt = 0; nt < 2; ++nt)
                acc[mt][nt] = __builtin_amdgcn_wmma_f32_16x16x32_bf16(
                    false, af[mt], false, bf[nt], (short)0, acc[mt][nt], false, false);
    };

    // ---- software pipeline ----
    load_tile(0);
    store_tile(0);
    __syncthreads();
    int buf = 0;
    for (int k0 = 32; k0 < Kdim; k0 += 32) {
        load_tile(k0);                                           // loads in flight...
        if (k0 + 32 < Kdim) __builtin_prefetch(aptr + k0 + 32, 0, 1);
        consume(buf);                                            // ...behind the WMMAs
        store_tile(buf ^ 1);
        __syncthreads();
        buf ^= 1;
    }
    consume(buf);

    // ---- epilogue: C layout VGPR r -> M = hi*8 + r, N = lr ----
#pragma unroll
    for (int mt = 0; mt < 2; ++mt) {
#pragma unroll
        for (int nt = 0; nt < 2; ++nt) {
            const int n = n0 + wn + nt * 16 + lr;
#pragma unroll
            for (int r = 0; r < 8; ++r) {
                const int m = m0 + wm + mt * 16 + hi * 8 + r;
                if (m < M && n < Nout) {
                    float v = acc[mt][nt][r] * alpha;
                    if (rowScale) v *= rowScale[m];
                    if (bias)     v += bias[n];
                    if (resid)    v += resid[(long long)m * ldres + n];
                    if (doRelu)   v = fmaxf(v, 0.0f);
                    C[(long long)m * ldc + n] = v;
                }
            }
        }
    }
}

// ---------------------------------------------------------------------------
// Beat encoder: conv1d(1->32,k5,s2,p2)+relu -> conv1d(32->64,k5,s2,p2)+relu
//               -> mean over time -> concat rr -> fc(66->256).
// One 256-thread workgroup per beat (grid = B*N = 4096).
// ---------------------------------------------------------------------------
__global__ void __launch_bounds__(256)
beat_encoder(const float* beats, const float* rr,
             const float* w1, const float* b1,
             const float* w2, const float* b2,
             const float* fcw, const float* fcb, float* E)
{
    __shared__ float xs[256];
    __shared__ float y1[32 * 128];
    __shared__ float msum[64];
    __shared__ float feat[66];

    const int beat = blockIdx.x;
    const int t = threadIdx.x;

    xs[t] = beats[(long long)beat * 256 + t];
    if (t < 64) msum[t] = 0.0f;
    __syncthreads();

    // conv1: 32 channels x 128 time
    for (int idx = t; idx < 32 * 128; idx += 256) {
        const int c = idx >> 7, tt = idx & 127;
        float acc = b1[c];
#pragma unroll
        for (int j = 0; j < 5; ++j) {
            const int p = 2 * tt + j - 2;
            if (p >= 0 && p < 256) acc += w1[c * 5 + j] * xs[p];
        }
        y1[idx] = fmaxf(acc, 0.0f);
    }
    __syncthreads();

    // conv2 (64 x 64) + accumulate mean over time
    for (int idx = t; idx < 64 * 64; idx += 256) {
        const int c = idx >> 6, tt = idx & 63;
        float acc = b2[c];
        for (int ci = 0; ci < 32; ++ci) {
            const float* w  = w2 + (c * 32 + ci) * 5;
            const float* yr = y1 + ci * 128;
#pragma unroll
            for (int j = 0; j < 5; ++j) {
                const int p = 2 * tt + j - 2;
                if (p >= 0 && p < 128) acc += w[j] * yr[p];
            }
        }
        atomicAdd(&msum[c], fmaxf(acc, 0.0f));
    }
    __syncthreads();

    if (t < 64) feat[t] = msum[t] * (1.0f / 64.0f);
    if (t == 0) { feat[64] = rr[beat * 2 + 0]; feat[65] = rr[beat * 2 + 1]; }
    __syncthreads();

    // fc: 66 -> 256 (one output per thread)
    float acc = fcb[t];
    for (int k = 0; k < 66; ++k) acc += fcw[t * 66 + k] * feat[k];
    E[(long long)beat * 256 + t] = acc;
}

// ---------------------------------------------------------------------------
// Fused residual + LayerNorm over D=256: Y = LN(X + H) * g + b  (Y may alias X)
// ---------------------------------------------------------------------------
__global__ void __launch_bounds__(256)
ln_residual(const float* X, const float* Hres, const float* g, const float* b, float* Y)
{
    __shared__ float red[256];
    const int row = blockIdx.x;
    const int i = threadIdx.x;
    const long long o = (long long)row * 256 + i;
    const float v = X[o] + Hres[o];

    red[i] = v; __syncthreads();
    for (int s = 128; s > 0; s >>= 1) { if (i < s) red[i] += red[i + s]; __syncthreads(); }
    const float mean = red[0] * (1.0f / 256.0f);
    __syncthreads();
    const float d = v - mean;
    red[i] = d * d; __syncthreads();
    for (int s = 128; s > 0; s >>= 1) { if (i < s) red[i] += red[i + s]; __syncthreads(); }
    const float var = red[0] * (1.0f / 256.0f);
    Y[o] = d * rsqrtf(var + 1e-5f) * g[i] + b[i];
}

// ---------------------------------------------------------------------------
// Row softmax (in place), one 256-thread block per row.
// ---------------------------------------------------------------------------
__global__ void __launch_bounds__(256)
softmax_rows(float* S, int ncols)
{
    __shared__ float red[256];
    float* p = S + (long long)blockIdx.x * ncols;
    const int t = threadIdx.x;

    float mx = -3.402823466e38f;
    for (int j = t; j < ncols; j += 256) mx = fmaxf(mx, p[j]);
    red[t] = mx; __syncthreads();
    for (int s = 128; s > 0; s >>= 1) { if (t < s) red[t] = fmaxf(red[t], red[t + s]); __syncthreads(); }
    mx = red[0]; __syncthreads();

    float sum = 0.0f;
    for (int j = t; j < ncols; j += 256) { float e = __expf(p[j] - mx); p[j] = e; sum += e; }
    red[t] = sum; __syncthreads();
    for (int s = 128; s > 0; s >>= 1) { if (t < s) red[t] += red[t + s]; __syncthreads(); }
    const float inv = 1.0f / red[0];
    for (int j = t; j < ncols; j += 256) p[j] *= inv;
}

// ---------------------------------------------------------------------------
// Graph build: per row top-5 of SIM -> ADJ row (0/1), OR in band(+-1), 1/deg.
// One block per (b, n) row; N = 1024.
// ---------------------------------------------------------------------------
__global__ void __launch_bounds__(256)
build_graph(const float* SIM, float* ADJ, float* DEGINV)
{
    __shared__ float red[256];
    const int row = blockIdx.x;           // b*N + i
    const int i   = row & 1023;
    const float* s = SIM + (long long)row * 1024;
    float* a = ADJ + (long long)row * 1024;
    const int t = threadIdx.x;

    for (int j = t; j < 1024; j += 256) a[j] = 0.0f;
    __syncthreads();

    if (t == 0) {
        float bv[5]; int bi[5];
#pragma unroll
        for (int q = 0; q < 5; ++q) { bv[q] = -3.402823466e38f; bi[q] = 0; }
        for (int j = 0; j < 1024; ++j) {
            const float v = s[j];
            if (v > bv[4]) {
                int q = 4;
                while (q > 0 && v > bv[q - 1]) { bv[q] = bv[q - 1]; bi[q] = bi[q - 1]; --q; }
                bv[q] = v; bi[q] = j;
            }
        }
#pragma unroll
        for (int q = 0; q < 5; ++q) a[bi[q]] = 1.0f;
        if (i > 0)    a[i - 1] = 1.0f;   // band eye(k=-1)
        if (i < 1023) a[i + 1] = 1.0f;   // band eye(k=+1)
    }
    __syncthreads();

    float sum = 0.0f;
    for (int j = t; j < 1024; j += 256) sum += a[j];
    red[t] = sum; __syncthreads();
    for (int s2 = 128; s2 > 0; s2 >>= 1) { if (t < s2) red[t] += red[t + s2]; __syncthreads(); }
    if (t == 0) DEGINV[row] = 1.0f / (red[0] + 1e-6f);
}

// ---------------------------------------------------------------------------
// EM = where(node_mask, mask_token, E) * valid_mask   (per-element, D=256 rows)
// ---------------------------------------------------------------------------
__global__ void __launch_bounds__(256)
mask_embed(const float* E, const int* nmask, const int* vmask, const float* mtok, float* EM)
{
    const long long i = (long long)blockIdx.x * blockDim.x + threadIdx.x;
    const long long row = i >> 8;
    const int d = (int)(i & 255);
    const float v = nmask[row] ? mtok[d] : E[i];
    EM[i] = vmask[row] ? v : 0.0f;
}

__global__ void __launch_bounds__(256)
copy_f32(const float* src, float* dst)
{
    const long long i = (long long)blockIdx.x * blockDim.x + threadIdx.x;
    dst[i] = src[i];
}

// ---------------------------------------------------------------------------
// Host orchestration
// ---------------------------------------------------------------------------
extern "C" void kernel_launch(void* const* d_in, const int* in_sizes, int n_in,
                              void* d_out, int out_size, void* d_ws, size_t ws_size,
                              hipStream_t stream)
{
    (void)in_sizes; (void)n_in; (void)out_size; (void)ws_size;

    const float* beats       = (const float*)d_in[0];
    const float* rr          = (const float*)d_in[1];
    const int*   node_mask   = (const int*)  d_in[2];
    const int*   valid_mask  = (const int*)  d_in[3];
    const float* conv1_w     = (const float*)d_in[4];
    const float* conv1_b     = (const float*)d_in[5];
    const float* conv2_w     = (const float*)d_in[6];
    const float* conv2_b     = (const float*)d_in[7];
    const float* fc_w        = (const float*)d_in[8];
    const float* fc_b        = (const float*)d_in[9];
    const float* mask_token  = (const float*)d_in[10];
    const float* tr_qkv_w    = (const float*)d_in[11];
    const float* tr_qkv_b    = (const float*)d_in[12];
    const float* tr_out_w    = (const float*)d_in[13];
    const float* tr_out_b    = (const float*)d_in[14];
    const float* tr_ln1_g    = (const float*)d_in[15];
    const float* tr_ln1_b    = (const float*)d_in[16];
    const float* tr_w1       = (const float*)d_in[17];
    const float* tr_b1       = (const float*)d_in[18];
    const float* tr_w2       = (const float*)d_in[19];
    const float* tr_b2       = (const float*)d_in[20];
    const float* tr_ln2_g    = (const float*)d_in[21];
    const float* tr_ln2_b    = (const float*)d_in[22];
    const float* gnn_self_w  = (const float*)d_in[23];
    const float* gnn_self_b  = (const float*)d_in[24];
    const float* gnn_neigh_w = (const float*)d_in[25];
    const float* gnn_neigh_b = (const float*)d_in[26];
    const float* dec_w       = (const float*)d_in[27];
    const float* dec_b       = (const float*)d_in[28];

    // ---- workspace arena (floats) ----
    float* W = (float*)d_ws;
    const long long MEG = 1048576LL;            // B*N*D = 4096*256
    float* E    = W + 0  * MEG;                 // embeddings           (4 MB)
    float* XT   = W + 1  * MEG;                 // target branch x      (4 MB)
    float* XM   = W + 2  * MEG;                 // masked branch x      (4 MB)
    float* QKV  = W + 3  * MEG;                 // (BN, 768)           (12 MB)
    float* Sbuf = W + 6  * MEG;                 // per-batch scores    (16 MB)
    float* HB   = W + 10 * MEG;                 // FFN hidden          (32 MB)
    float* TMP  = W + 18 * MEG;                 // attn merged / self   (4 MB)
    float* TMP2 = W + 19 * MEG;                 // proj / ffn out       (4 MB)
    float* SIM  = W + 20 * MEG;                 // x @ x^T             (16 MB)
    float* ADJ  = W + 24 * MEG;                 // adjacency           (16 MB)
    float* AGG  = W + 28 * MEG;                 // GNN aggregate        (4 MB)
    float* DEGI = W + 29 * MEG;                 // 1/deg               (16 KB)

    auto gemm = [&](const float* Ap, int lda, long long aB,
                    const float* Bp, int ldb, long long bB, int bNK,
                    const float* bias,
                    const float* resid, int ldres, long long resB,
                    const float* rs, long long rsB,
                    float* Cp, int ldc, long long cB,
                    int M, int Nc, int Kd, int relu, float alpha, int batch) {
        dim3 g((unsigned)((Nc + 63) / 64), (unsigned)((M + 63) / 64), (unsigned)batch);
        gemm_bf16_wmma<<<g, dim3(128), 0, stream>>>(
            Ap, lda, aB, Bp, ldb, bB, bNK, bias, resid, ldres, resB,
            rs, rsB, Cp, ldc, cB, M, Nc, Kd, relu, alpha);
    };

    // ---- 2-layer post-norm transformer (in-place on X) ----
    auto temporal = [&](float* X) {
        for (int l = 0; l < 2; ++l) {
            // QKV = X @ qkv_w^T + qkv_b        (4096 x 768, K=256)
            gemm(X, 256, 0,
                 tr_qkv_w + (long long)l * 768 * 256, 256, 0, 1,
                 tr_qkv_b + (long long)l * 768,
                 nullptr, 0, 0, nullptr, 0,
                 QKV, 768, 0, 4096, 768, 256, 0, 1.0f, 1);
            for (int b = 0; b < 4; ++b) {
                const float* qkvb = QKV + (long long)b * 1024 * 768;
                // S = (Q @ K^T) / sqrt(dh)     (z = head, stride 64 within 768)
                gemm(qkvb, 768, 64,
                     qkvb + 256, 768, 64, 1,
                     nullptr, nullptr, 0, 0, nullptr, 0,
                     Sbuf, 1024, 1024LL * 1024, 1024, 1024, 64, 0, 0.125f, 4);
                softmax_rows<<<4 * 1024, 256, 0, stream>>>(Sbuf, 1024);
                // O = P @ V, merge heads into (N, 256) at column h*64
                gemm(Sbuf, 1024, 1024LL * 1024,
                     qkvb + 512, 768, 64, 0,
                     nullptr, nullptr, 0, 0, nullptr, 0,
                     TMP + (long long)b * 1024 * 256, 256, 64, 1024, 64, 1024, 0, 1.0f, 4);
            }
            // TMP2 = O @ out_w^T + out_b ;  X = LN(X + TMP2)
            gemm(TMP, 256, 0,
                 tr_out_w + (long long)l * 256 * 256, 256, 0, 1,
                 tr_out_b + (long long)l * 256,
                 nullptr, 0, 0, nullptr, 0,
                 TMP2, 256, 0, 4096, 256, 256, 0, 1.0f, 1);
            ln_residual<<<4096, 256, 0, stream>>>(X, TMP2,
                                                  tr_ln1_g + l * 256, tr_ln1_b + l * 256, X);
            // FFN: HB = relu(X @ w1^T + b1) ; TMP2 = HB @ w2^T + b2 ; X = LN(X + TMP2)
            gemm(X, 256, 0,
                 tr_w1 + (long long)l * 2048 * 256, 256, 0, 1,
                 tr_b1 + (long long)l * 2048,
                 nullptr, 0, 0, nullptr, 0,
                 HB, 2048, 0, 4096, 2048, 256, 1, 1.0f, 1);
            gemm(HB, 2048, 0,
                 tr_w2 + (long long)l * 256 * 2048, 2048, 0, 1,
                 tr_b2 + (long long)l * 256,
                 nullptr, 0, 0, nullptr, 0,
                 TMP2, 256, 0, 4096, 256, 2048, 0, 1.0f, 1);
            ln_residual<<<4096, 256, 0, stream>>>(X, TMP2,
                                                  tr_ln2_g + l * 256, tr_ln2_b + l * 256, X);
        }
    };

    // ---- 1) beat encoder -> E ----
    beat_encoder<<<4096, 256, 0, stream>>>(beats, rr, conv1_w, conv1_b,
                                           conv2_w, conv2_b, fc_w, fc_b, E);

    // ---- 2) branches ----
    copy_f32<<<4096, 256, 0, stream>>>(E, XT);
    mask_embed<<<4096, 256, 0, stream>>>(E, node_mask, valid_mask, mask_token, XM);
    temporal(XT);   // target branch (stop_gradient is a no-op in forward)
    temporal(XM);   // masked branch

    // ---- 3) graph build: SIM = XM @ XM^T per batch; top-5 + band; 1/deg ----
    gemm(XM, 256, 1024LL * 256,
         XM, 256, 1024LL * 256, 1,
         nullptr, nullptr, 0, 0, nullptr, 0,
         SIM, 1024, 1024LL * 1024, 1024, 1024, 256, 0, 1.0f, 4);
    build_graph<<<4096, 256, 0, stream>>>(SIM, ADJ, DEGI);

    // ---- 4) 3-layer GNN (in place on XM) ----
    for (int l = 0; l < 3; ++l) {
        // AGG = (ADJ @ XM) / deg   (per batch)
        gemm(ADJ, 1024, 1024LL * 1024,
             XM, 256, 1024LL * 256, 0,
             nullptr, nullptr, 0, 0,
             DEGI, 1024,
             AGG, 256, 1024LL * 256, 1024, 256, 1024, 0, 1.0f, 4);
        // TMP = XM @ self_w^T + self_b
        gemm(XM, 256, 0,
             gnn_self_w + (long long)l * 256 * 256, 256, 0, 1,
             gnn_self_b + (long long)l * 256,
             nullptr, 0, 0, nullptr, 0,
             TMP, 256, 0, 4096, 256, 256, 0, 1.0f, 1);
        // XM = AGG @ neigh_w^T + neigh_b + TMP
        gemm(AGG, 256, 0,
             gnn_neigh_w + (long long)l * 256 * 256, 256, 0, 1,
             gnn_neigh_b + (long long)l * 256,
             TMP, 256, 0, nullptr, 0,
             XM, 256, 0, 4096, 256, 256, 0, 1.0f, 1);
    }

    // ---- 5) outputs: [target | recon] ----
    float* out = (float*)d_out;
    copy_f32<<<4096, 256, 0, stream>>>(XT, out);
    gemm(XM, 256, 0,
         dec_w, 256, 0, 1,
         dec_b,
         nullptr, 0, 0, nullptr, 0,
         out + 1048576, 256, 0, 4096, 256, 256, 0, 1.0f, 1);
}
